// TransducerPrediction_3582002725422
// MI455X (gfx1250) — compile-verified, matrix operand order checked by reading, not run on hardware
//
#include <hip/hip_runtime.h>
#include <hip/hip_bf16.h>

// MI455X / gfx1250, wave32. bf16 WMMA with f32 accumulate.

typedef __attribute__((ext_vector_type(16))) __bf16 v16bf;
typedef __attribute__((ext_vector_type(8)))  float  v8f;
typedef __attribute__((ext_vector_type(4)))  int    v4i;

typedef __attribute__((address_space(1))) v4i* gptr4;   // global int32x4*
typedef __attribute__((address_space(3))) v4i* sptr4;   // LDS int32x4*

#define VOCAB 1024
#define EMBED 512
#define UNITS 1024
#define BATCH 64
#define SEQT  512
#define NGATE 4096   // 4*UNITS
#define SCAN_BLOCKS 32

#if defined(__AMDGCN__) && __has_builtin(__builtin_amdgcn_global_load_async_to_lds_b128) && \
    __has_builtin(__builtin_amdgcn_s_wait_asynccnt)
#define HAVE_ASYNC_LDS 1
#else
#define HAVE_ASYNC_LDS 0
#endif

// Copy 32 bytes (16 bf16) global -> LDS. Async (no VGPR data, ASYNCcnt) when available.
__device__ __forceinline__ void cp32_g2s(const __bf16* g, __bf16* s) {
#if HAVE_ASYNC_LDS
  __builtin_amdgcn_global_load_async_to_lds_b128(
      (gptr4)const_cast<__bf16*>(g), (sptr4)s, 0, 0);
  __builtin_amdgcn_global_load_async_to_lds_b128(
      (gptr4)const_cast<__bf16*>(g + 8), (sptr4)(s + 8), 0, 0);
#else
  *(v16bf*)s = *(const v16bf*)g;
#endif
}

__device__ __forceinline__ void wait_stage() {
#if HAVE_ASYNC_LDS
  __builtin_amdgcn_s_wait_asynccnt(0);
#endif
  __syncthreads();
}

// ---------------- embedding gather + f32->bf16 cast ----------------
__global__ void k_embed_cast(const int* __restrict__ tok,
                             const float* __restrict__ tab,
                             __bf16* __restrict__ x) {
  int bt = blockIdx.x;                       // bt = b*T + t
  int t0 = tok[bt];
  const float* row = tab + (size_t)t0 * EMBED;
  __bf16* o = x + (size_t)bt * EMBED;
  for (int d = threadIdx.x; d < EMBED; d += 256) o[d] = (__bf16)row[d];
}

// ---------------- weight cast + transpose: w[K,N] f32 -> wt[N,K] bf16 ----------------
__global__ void k_cast_transpose(const float* __restrict__ w,
                                 __bf16* __restrict__ wt, int K, int N) {
  long idx = (long)blockIdx.x * 256 + threadIdx.x;   // over N*K
  int k = (int)(idx % K);
  int n = (int)(idx / K);
  wt[(size_t)n * K + k] = (__bf16)w[(size_t)k * N + n];
}

// ---------------- big GEMM: C[M,N] = A[M,K](bf16) * Bt[N,K](bf16)^T + bias ----------------
// Workgroup tile 128x128, double-buffered LDS staging (async-to-LDS when available),
// 8 waves; wave tile 64x32 -> 4x2 wmma accumulators.
__global__ __launch_bounds__(256) void k_gemm_bf16(const __bf16* __restrict__ A,
                                                   const __bf16* __restrict__ Bt,
                                                   const float* __restrict__ bias,
                                                   float* __restrict__ C,
                                                   int M, int N, int K) {
  __shared__ __align__(32) __bf16 As[2][128 * 32];
  __shared__ __align__(32) __bf16 Bs[2][128 * 32];   // stored [n][k]

  int tm = blockIdx.x, tn = blockIdx.y;
  int tid = threadIdx.x, lane = tid & 31, wid = tid >> 5;
  int wm = wid >> 2;            // 0..1  -> M offset wm*64
  int wn = wid & 3;             // 0..3  -> N offset wn*32
  int lm = lane & 15;           // fragment row/col within 16
  int lk = (lane >> 4) << 4;    // K half select (0 or 16)

  v8f acc[4][2] = {};

  int rowS = tid >> 1;                 // 0..127 staging row
  int kc   = (tid & 1) << 4;           // 0 or 16
  int sOff = rowS * 32 + kc;
  const __bf16* Ap = A  + (size_t)(tm * 128 + rowS) * K + kc;
  const __bf16* Bp = Bt + (size_t)(tn * 128 + rowS) * K + kc;

  // prologue: stage K-tile 0 into buffer 0
  cp32_g2s(Ap, &As[0][sOff]);
  cp32_g2s(Bp, &Bs[0][sOff]);

  int buf = 0;
  for (int k0 = 0; k0 < K; k0 += 32) {
    wait_stage();                      // staged data visible to all waves

    if (k0 + 32 < K) {                 // stage next tile into other buffer
      cp32_g2s(Ap + k0 + 32, &As[buf ^ 1][sOff]);
      cp32_g2s(Bp + k0 + 32, &Bs[buf ^ 1][sOff]);
      if (k0 + 64 < K) {
        __builtin_prefetch(Ap + k0 + 64, 0, 3);
        __builtin_prefetch(Bp + k0 + 64, 0, 3);
      }
    }

    v16bf b0 = *(const v16bf*)&Bs[buf][(wn * 32 +  0 + lm) * 32 + lk];
    v16bf b1 = *(const v16bf*)&Bs[buf][(wn * 32 + 16 + lm) * 32 + lk];
#pragma unroll
    for (int mt = 0; mt < 4; ++mt) {
      v16bf a = *(const v16bf*)&As[buf][(wm * 64 + mt * 16 + lm) * 32 + lk];
      acc[mt][0] = __builtin_amdgcn_wmma_f32_16x16x32_bf16(
          false, a, false, b0, (short)0, acc[mt][0], false, false);
      acc[mt][1] = __builtin_amdgcn_wmma_f32_16x16x32_bf16(
          false, a, false, b1, (short)0, acc[mt][1], false, false);
    }
    buf ^= 1;
  }

#pragma unroll
  for (int nt = 0; nt < 2; ++nt) {
    int col = tn * 128 + wn * 32 + nt * 16 + lm;
    float bv = bias ? bias[col] : 0.f;
#pragma unroll
    for (int mt = 0; mt < 4; ++mt) {
      int rbase = tm * 128 + wm * 64 + mt * 16 + ((lane >> 4) << 3);
#pragma unroll
      for (int r = 0; r < 8; ++r)
        C[(size_t)(rbase + r) * N + col] = acc[mt][nt][r] + bv;
    }
  }
}

// ---------------- persistent LSTM recurrence ----------------
__device__ __forceinline__ float sigm_(float x) { return 1.f / (1.f + __expf(-x)); }
__device__ __forceinline__ float tanh_(float x) {
  float e = __expf(2.f * x);
  return (e - 1.f) / (e + 1.f);
}

// 32 blocks x 8 waves = 256 waves; wave wg: group=wg/4 owns 16 units (cols),
// mt=wg%4 owns one 16-row M tile of the 64-row batch. c state lives in VGPRs.
__global__ __launch_bounds__(256) void k_lstm_scan(const __bf16* __restrict__ Ut,  // [4096][1024] bf16
                                                   const float* __restrict__ xz,   // [B*T][4096]
                                                   __bf16* __restrict__ hb,        // 2 x [64][1024] ping-pong
                                                   __bf16* __restrict__ outb,      // bf16 out (layer0) or null
                                                   float* __restrict__ outf,       // f32 out (layer1) or null
                                                   unsigned* __restrict__ bar) {
  int tid = threadIdx.x, lane = tid & 31, wid = tid >> 5;
  int wg = blockIdx.x * 8 + wid;
  int group = wg >> 2;                 // 0..63 -> unit column group
  int mt    = wg & 3;                  // 0..3  -> M tile
  int unit  = group * 16;
  int lm = lane & 15;
  int lk = (lane >> 4) << 4;
  int col = unit + lm;
  int rbase = mt * 16 + ((lane >> 4) << 3);

  const __bf16* bi = Ut + (size_t)(0 * UNITS + col) * UNITS + lk;
  const __bf16* bf = Ut + (size_t)(1 * UNITS + col) * UNITS + lk;
  const __bf16* bg = Ut + (size_t)(2 * UNITS + col) * UNITS + lk;
  const __bf16* bo = Ut + (size_t)(3 * UNITS + col) * UNITS + lk;

  v8f cst = {};

  for (int t = 0; t < SEQT; ++t) {
    const __bf16* hc = hb + (size_t)(t & 1) * (BATCH * UNITS);
    __bf16*       hn = hb + (size_t)((t + 1) & 1) * (BATCH * UNITS);

    v8f zi = {}, zf = {}, zg = {}, zo = {};
    const __bf16* arow = hc + (size_t)(mt * 16 + lm) * UNITS + lk;
#pragma unroll 4
    for (int k0 = 0; k0 < UNITS; k0 += 32) {
      v16bf a = *(const v16bf*)(arow + k0);
      zi = __builtin_amdgcn_wmma_f32_16x16x32_bf16(
          false, a, false, *(const v16bf*)(bi + k0), (short)0, zi, false, false);
      zf = __builtin_amdgcn_wmma_f32_16x16x32_bf16(
          false, a, false, *(const v16bf*)(bf + k0), (short)0, zf, false, false);
      zg = __builtin_amdgcn_wmma_f32_16x16x32_bf16(
          false, a, false, *(const v16bf*)(bg + k0), (short)0, zg, false, false);
      zo = __builtin_amdgcn_wmma_f32_16x16x32_bf16(
          false, a, false, *(const v16bf*)(bo + k0), (short)0, zo, false, false);
    }

#pragma unroll
    for (int r = 0; r < 8; ++r) {
      int m = rbase + r;                                   // batch row 0..63
      size_t xr = ((size_t)m * SEQT + t) * NGATE;
      float I = sigm_(zi[r] + xz[xr + 0 * UNITS + col]);
      float F = sigm_(zf[r] + xz[xr + 1 * UNITS + col]);
      float G = tanh_(zg[r] + xz[xr + 2 * UNITS + col]);
      float O = sigm_(zo[r] + xz[xr + 3 * UNITS + col]);
      float cn = F * cst[r] + I * G;
      cst[r] = cn;
      float h = O * tanh_(cn);
      hn[(size_t)m * UNITS + col] = (__bf16)h;
      size_t orow = ((size_t)m * SEQT + t) * UNITS + col;
      if (outb) outb[orow] = (__bf16)h;
      if (outf) outf[orow] = h;
    }

    // grid-wide barrier: all 32 resident blocks must see h before next step
    __syncthreads();
    if (tid == 0) {
      __threadfence();
      atomicAdd(bar, 1u);
      unsigned target = (unsigned)(SCAN_BLOCKS * (t + 1));
      while (atomicAdd(bar, 0u) < target) { __builtin_amdgcn_s_sleep(1); }
      __threadfence();
    }
    __syncthreads();
  }
}

extern "C" void kernel_launch(void* const* d_in, const int* in_sizes, int n_in,
                              void* d_out, int out_size, void* d_ws, size_t ws_size,
                              hipStream_t stream) {
  const int*   tok = (const int*)d_in[0];
  const float* tab = (const float*)d_in[1];
  const float* W0  = (const float*)d_in[2];
  const float* U0  = (const float*)d_in[3];
  const float* b0  = (const float*)d_in[4];
  const float* W1  = (const float*)d_in[5];
  const float* U1  = (const float*)d_in[6];
  const float* b1  = (const float*)d_in[7];
  float* out = (float*)d_out;

  char* w = (char*)d_ws;
  size_t o = 0;
  auto alloc = [&](size_t bytes) -> char* {
    char* p = w + o;
    o += (bytes + 255) & ~(size_t)255;
    return p;
  };
  float*  xz  = (float*) alloc((size_t)BATCH * SEQT * NGATE * 4);   // 512 MB gate preacts
  __bf16* x0  = (__bf16*)alloc((size_t)BATCH * SEQT * EMBED * 2);   // embeddings bf16
  __bf16* x1  = (__bf16*)alloc((size_t)BATCH * SEQT * UNITS * 2);   // layer0 output bf16
  __bf16* W0t = (__bf16*)alloc((size_t)NGATE * EMBED * 2);
  __bf16* U0t = (__bf16*)alloc((size_t)NGATE * UNITS * 2);
  __bf16* W1t = (__bf16*)alloc((size_t)NGATE * UNITS * 2);
  __bf16* U1t = (__bf16*)alloc((size_t)NGATE * UNITS * 2);
  __bf16* hb  = (__bf16*)alloc((size_t)2 * BATCH * UNITS * 2);      // h ping-pong
  unsigned* bar = (unsigned*)alloc(256);

  // 1) embedding gather + casts/transposes (one-time, ~180 MB read+written)
  k_embed_cast<<<BATCH * SEQT, 256, 0, stream>>>(tok, tab, x0);
  k_cast_transpose<<<(EMBED * NGATE) / 256, 256, 0, stream>>>(W0, W0t, EMBED, NGATE);
  k_cast_transpose<<<(UNITS * NGATE) / 256, 256, 0, stream>>>(U0, U0t, UNITS, NGATE);
  k_cast_transpose<<<(UNITS * NGATE) / 256, 256, 0, stream>>>(W1, W1t, UNITS, NGATE);
  k_cast_transpose<<<(UNITS * NGATE) / 256, 256, 0, stream>>>(U1, U1t, UNITS, NGATE);

  dim3 gg(BATCH * SEQT / 128, NGATE / 128);

  // 2) layer 0: xz0 = x0 @ W0 + b0, then sequential scan
  k_gemm_bf16<<<gg, 256, 0, stream>>>(x0, W0t, b0, xz, BATCH * SEQT, NGATE, EMBED);
  (void)hipMemsetAsync(hb, 0, (size_t)2 * BATCH * UNITS * 2, stream);
  (void)hipMemsetAsync(bar, 0, 256, stream);
  k_lstm_scan<<<SCAN_BLOCKS, 256, 0, stream>>>(U0t, xz, hb, x1, nullptr, bar);

  // 3) layer 1: xz1 = x1 @ W1 + b1, then scan writing f32 output
  k_gemm_bf16<<<gg, 256, 0, stream>>>(x1, W1t, b1, xz, BATCH * SEQT, NGATE, UNITS);
  (void)hipMemsetAsync(hb, 0, (size_t)2 * BATCH * UNITS * 2, stream);
  (void)hipMemsetAsync(bar, 0, 256, stream);
  k_lstm_scan<<<SCAN_BLOCKS, 256, 0, stream>>>(U1t, xz, hb, nullptr, out, bar);
}
